// PressLightNet_90950227460697
// MI455X (gfx1250) — matrix operand order repaired and checked by reading, not compiled, not run on hardware
//
#include <hip/hip_runtime.h>

typedef __attribute__((ext_vector_type(16))) __bf16 v16bf;
typedef __attribute__((ext_vector_type(8)))  float  v8f;

#define B_ROWS   32768
#define P_NUM    8
#define PHASE_W  8
#define FEAT_W   56
#define IN_DIM   64
#define HDIM     1024
#define A_DIM    8

// ---------------- workspace layout (bytes) ----------------
static constexpr size_t OFF_CNT  = 0;                                   // 8 int
static constexpr size_t OFF_CNT2 = 32;                                  // 8 int
static constexpr size_t OFF_OFFS = 64;                                  // 8 int
static constexpr size_t OFF_IDX  = 256;                                 // B ints
static constexpr size_t OFF_PERM = OFF_IDX  + (size_t)B_ROWS * 4;       // B ints
static constexpr size_t OFF_W0T  = OFF_PERM + (size_t)B_ROWS * 4;       // [n][k] bf16
static constexpr size_t OFF_W1T  = OFF_W0T  + (size_t)IN_DIM * HDIM * 2;       // [p][n][k] bf16
static constexpr size_t OFF_W2T  = OFF_W1T  + (size_t)P_NUM * HDIM * HDIM * 2; // [p][n][k] bf16 (n<8)
static constexpr size_t OFF_S    = OFF_W2T  + (size_t)P_NUM * A_DIM * HDIM * 2; // [B][H] bf16

__device__ __forceinline__ unsigned short f2bf(float f) {
    unsigned int u = __float_as_uint(f);
    u += 0x7FFFu + ((u >> 16) & 1u);           // round-to-nearest-even truncation
    return (unsigned short)(u >> 16);
}

union Frag16 { v16bf v; unsigned int u[8]; uint4 q[2]; };

// ---------------- bucketing ----------------
__global__ void zero_counters_kernel(int* c) {
    if (threadIdx.x < 24) c[threadIdx.x] = 0;   // cnt[8], cnt2[8], offs[8]
}

__global__ void phase_count_kernel(const float* __restrict__ cur_phase,
                                   int* __restrict__ idx, int* __restrict__ cnt) {
    int b = blockIdx.x * blockDim.x + threadIdx.x;
    if (b >= B_ROWS) return;
    int p = 0;
    #pragma unroll
    for (int j = PHASE_W - 1; j >= 0; --j)
        if (cur_phase[b * PHASE_W + j] > 0.5f) p = j;   // one-hot argmax, default 0
    idx[b] = p;
    atomicAdd(&cnt[p], 1);
}

__global__ void phase_scan_kernel(const int* __restrict__ cnt, int* __restrict__ offs) {
    if (threadIdx.x == 0) {
        int acc = 0;
        for (int p = 0; p < P_NUM; ++p) { offs[p] = acc; acc += cnt[p]; }
    }
}

__global__ void phase_scatter_kernel(const int* __restrict__ idx, const int* __restrict__ offs,
                                     int* __restrict__ cnt2, int* __restrict__ perm) {
    int b = blockIdx.x * blockDim.x + threadIdx.x;
    if (b >= B_ROWS) return;
    int p = idx[b];
    int pos = offs[p] + atomicAdd(&cnt2[p], 1);
    perm[pos] = b;
}

// ---------------- weight convert/transpose to bf16, K-contiguous ----------------
__global__ void cvt_w0_kernel(const float* __restrict__ W0, unsigned short* __restrict__ w0t) {
    int i = blockIdx.x * 256 + threadIdx.x;        // 64*1024
    int n = i >> 6, k = i & 63;
    w0t[n * IN_DIM + k] = f2bf(W0[(size_t)k * HDIM + n]);
}

__global__ void cvt_w1_kernel(const float* __restrict__ W1, unsigned short* __restrict__ w1t) {
    size_t i = (size_t)blockIdx.x * 256 + threadIdx.x;   // 8*1024*1024
    int p = (int)(i >> 20);
    int n = (int)((i >> 10) & 1023);
    int k = (int)(i & 1023);
    w1t[i] = f2bf(W1[(size_t)p * HDIM * HDIM + (size_t)k * HDIM + n]);
}

__global__ void cvt_w2_kernel(const float* __restrict__ W2, unsigned short* __restrict__ w2t) {
    int i = blockIdx.x * 256 + threadIdx.x;        // 8*8*1024
    int p = i >> 13;
    int n = (i >> 10) & 7;
    int k = i & 1023;
    w2t[i] = f2bf(W2[(size_t)p * HDIM * A_DIM + (size_t)k * A_DIM + n]);
}

// ---------------- layer 0: x[B,64] @ W0[64,1024] -> sigmoid -> s (bf16) ----------------
__device__ __forceinline__ float load_x(const float* __restrict__ cp,
                                        const float* __restrict__ ft, int row, int k) {
    return (k < PHASE_W) ? cp[row * PHASE_W + k] : ft[row * FEAT_W + (k - PHASE_W)];
}

__global__ __launch_bounds__(256) void layer0_kernel(
    const float* __restrict__ cur_phase, const float* __restrict__ feat,
    const unsigned short* __restrict__ w0t, const float* __restrict__ b0,
    unsigned short* __restrict__ sbuf) {

    const int mt = blockIdx.x;              // 64-row M-block
    __shared__ __align__(16) unsigned short xs[64 * IN_DIM];   // 8 KB: x tile bf16

    {   // build x tile in LDS (each thread: one row quarter = 16 elems)
        int r = threadIdx.x >> 2;
        int q = (threadIdx.x & 3) * 16;
        int grow = mt * 64 + r;
        #pragma unroll
        for (int t = 0; t < 16; ++t)
            xs[r * IN_DIM + q + t] = f2bf(load_x(cur_phase, feat, grow, q + t));
    }
    __syncthreads();

    const int wave = threadIdx.x >> 5;
    const int lane = threadIdx.x & 31;
    const int lrow = lane & 15;
    const int half = lane >> 4;
    const int akb = half * 8;
    const int bkb = half * 8;

    #pragma unroll 1
    for (int j = 0; j < 8; ++j) {
        int n = (wave * 8 + j) * 16 + lrow;
        const unsigned short* bcol = w0t + (size_t)n * IN_DIM;
        v8f acc[4] = {};
        #pragma unroll
        for (int kk = 0; kk < 2; ++kk) {
            int kOff = kk * 32;
            Frag16 bf;
            bf.q[0] = *(const uint4*)(bcol + kOff + bkb);
            bf.q[1] = *(const uint4*)(bcol + kOff + bkb + 16);
            #pragma unroll
            for (int ms = 0; ms < 4; ++ms) {
                Frag16 af;
                const unsigned short* arow = xs + (ms * 16 + lrow) * IN_DIM;
                af.q[0] = *(const uint4*)(arow + kOff + akb);
                af.q[1] = *(const uint4*)(arow + kOff + akb + 16);
                acc[ms] = __builtin_amdgcn_wmma_f32_16x16x32_bf16(
                    false, af.v, false, bf.v, (short)0, acc[ms], false, false);
            }
        }
        float bias = b0[n];
        #pragma unroll
        for (int ms = 0; ms < 4; ++ms) {
            #pragma unroll
            for (int r = 0; r < 8; ++r) {
                int m = ms * 16 + r + half * 8;
                float v = acc[ms][r] + bias;
                float sg = __builtin_amdgcn_rcpf(1.0f + __expf(-v));  // fast sigmoid
                sbuf[(size_t)(mt * 64 + m) * HDIM + n] = f2bf(sg);
            }
        }
    }
}

// ---------------- fused layer 1 + layer 2, per phase bucket (32 rows/block) ----------------
__global__ __launch_bounds__(256) void layer12_kernel(
    const unsigned short* __restrict__ sbuf, const unsigned short* __restrict__ w1t,
    const float* __restrict__ b1, const unsigned short* __restrict__ w2t,
    const float* __restrict__ b2, const int* __restrict__ cnt,
    const int* __restrict__ offs, const int* __restrict__ perm,
    float* __restrict__ out) {

    const int p    = blockIdx.y;
    const int tile = blockIdx.x;
    const int count = cnt[p];
    if (tile * 32 >= count) return;                 // uniform early exit
    const int base = offs[p];

    __shared__ __align__(16) unsigned short sA[32 * HDIM];  // 64 KB gathered s rows
    __shared__ __align__(16) unsigned short sH[32 * HDIM];  // 64 KB hidden tile
    __shared__ int rows[32];

    if (threadIdx.x < 32) {
        int slot = tile * 32 + threadIdx.x;
        int cs = slot < count ? slot : count - 1;
        rows[threadIdx.x] = perm[base + cs];
    }
    __syncthreads();

    {   // cooperative 16B-vector gather: 8 threads per row, 128 ushorts each
        int r  = threadIdx.x >> 3;
        int c0 = (threadIdx.x & 7) * 128;
        const uint4* src = (const uint4*)(sbuf + (size_t)rows[r] * HDIM + c0);
        uint4* dst = (uint4*)(sA + r * HDIM + c0);
        #pragma unroll
        for (int t = 0; t < 16; ++t) dst[t] = src[t];
    }
    __syncthreads();

    const int wave = threadIdx.x >> 5;
    const int lane = threadIdx.x & 31;
    const int lrow = lane & 15;
    const int half = lane >> 4;
    const int akb = half * 8;
    const int bkb = half * 8;

    const unsigned short* w1p = w1t + (size_t)p * HDIM * HDIM;

    // ---- layer 1: h[32,1024] = relu(sA @ W1[p] + b1[p]); B-frag reused for 2 M-subtiles
    #pragma unroll 1
    for (int j = 0; j < 8; ++j) {
        int n = (wave * 8 + j) * 16 + lrow;
        const unsigned short* bcol = w1p + (size_t)n * HDIM;
        // prefetch next column for this wave (gfx1250 global_prefetch_b8)
        __builtin_prefetch(w1p + (size_t)((wave * 8 + ((j + 1) & 7)) * 16 + lrow) * HDIM, 0, 1);
        v8f acc0 = {}, acc1 = {};
        const unsigned short* arow0 = sA + lrow * HDIM;
        const unsigned short* arow1 = sA + (16 + lrow) * HDIM;
        #pragma unroll 2
        for (int kk = 0; kk < 32; ++kk) {
            int kOff = kk * 32;
            Frag16 bf, af0, af1;
            bf.q[0]  = *(const uint4*)(bcol + kOff + bkb);
            bf.q[1]  = *(const uint4*)(bcol + kOff + bkb + 16);
            af0.q[0] = *(const uint4*)(arow0 + kOff + akb);
            af0.q[1] = *(const uint4*)(arow0 + kOff + akb + 16);
            af1.q[0] = *(const uint4*)(arow1 + kOff + akb);
            af1.q[1] = *(const uint4*)(arow1 + kOff + akb + 16);
            acc0 = __builtin_amdgcn_wmma_f32_16x16x32_bf16(
                false, af0.v, false, bf.v, (short)0, acc0, false, false);
            acc1 = __builtin_amdgcn_wmma_f32_16x16x32_bf16(
                false, af1.v, false, bf.v, (short)0, acc1, false, false);
        }
        float bias = b1[p * HDIM + n];
        #pragma unroll
        for (int r = 0; r < 8; ++r) {
            int m = r + half * 8;
            float v0 = acc0[r] + bias;
            float v1 = acc1[r] + bias;
            sH[m * HDIM + n]        = f2bf(v0 > 0.0f ? v0 : 0.0f);
            sH[(16 + m) * HDIM + n] = f2bf(v1 > 0.0f ? v1 : 0.0f);
        }
    }
    __syncthreads();

    // ---- layer 2: out[32,8] = h @ W2[p] + b2[p]; waves 0/1 take one 16-row subtile each
    if (wave < 2) {
        const int ms = wave;
        const unsigned short* w2p = w2t + (size_t)p * A_DIM * HDIM;
        const unsigned short* arow = sH + (ms * 16 + lrow) * HDIM;
        const int n = lrow;                 // cols 8..15 are zero pad
        v8f acc = {};
        #pragma unroll 2
        for (int kk = 0; kk < 32; ++kk) {
            int kOff = kk * 32;
            Frag16 af, bf;
            af.q[0] = *(const uint4*)(arow + kOff + akb);
            af.q[1] = *(const uint4*)(arow + kOff + akb + 16);
            if (n < A_DIM) {
                const unsigned short* bcol = w2p + (size_t)n * HDIM;
                bf.q[0] = *(const uint4*)(bcol + kOff + bkb);
                bf.q[1] = *(const uint4*)(bcol + kOff + bkb + 16);
            } else {
                bf.q[0] = make_uint4(0u, 0u, 0u, 0u);
                bf.q[1] = make_uint4(0u, 0u, 0u, 0u);
            }
            acc = __builtin_amdgcn_wmma_f32_16x16x32_bf16(
                false, af.v, false, bf.v, (short)0, acc, false, false);
        }
        if (n < A_DIM) {
            float bias = b2[p * A_DIM + n];
            #pragma unroll
            for (int r = 0; r < 8; ++r) {
                int m = r + half * 8;
                int slot = tile * 32 + ms * 16 + m;
                if (slot < count) {
                    int grow = perm[base + slot];
                    out[(size_t)grow * A_DIM + n] = acc[r] + bias;
                }
            }
        }
    }
}

extern "C" void kernel_launch(void* const* d_in, const int* in_sizes, int n_in,
                              void* d_out, int out_size, void* d_ws, size_t ws_size,
                              hipStream_t stream) {
    (void)in_sizes; (void)n_in; (void)out_size; (void)ws_size;
    const float* cur_phase = (const float*)d_in[0];
    const float* feat      = (const float*)d_in[1];
    const float* W0        = (const float*)d_in[2];
    const float* b0        = (const float*)d_in[3];
    const float* W1        = (const float*)d_in[4];
    const float* b1        = (const float*)d_in[5];
    const float* W2        = (const float*)d_in[6];
    const float* b2        = (const float*)d_in[7];
    // d_in[8] phase_table: identity mapping, handled by argmax

    float* out = (float*)d_out;
    char* ws = (char*)d_ws;
    int* cnt   = (int*)(ws + OFF_CNT);
    int* cnt2  = (int*)(ws + OFF_CNT2);
    int* offs  = (int*)(ws + OFF_OFFS);
    int* idx   = (int*)(ws + OFF_IDX);
    int* perm  = (int*)(ws + OFF_PERM);
    unsigned short* w0t  = (unsigned short*)(ws + OFF_W0T);
    unsigned short* w1t  = (unsigned short*)(ws + OFF_W1T);
    unsigned short* w2t  = (unsigned short*)(ws + OFF_W2T);
    unsigned short* sbuf = (unsigned short*)(ws + OFF_S);

    zero_counters_kernel<<<1, 32, 0, stream>>>(cnt);
    phase_count_kernel<<<B_ROWS / 256, 256, 0, stream>>>(cur_phase, idx, cnt);
    phase_scan_kernel<<<1, 1, 0, stream>>>(cnt, offs);
    phase_scatter_kernel<<<B_ROWS / 256, 256, 0, stream>>>(idx, offs, cnt2, perm);

    cvt_w0_kernel<<<(IN_DIM * HDIM) / 256, 256, 0, stream>>>(W0, w0t);
    cvt_w1_kernel<<<(P_NUM * HDIM * HDIM) / 256, 256, 0, stream>>>(W1, w1t);
    cvt_w2_kernel<<<(P_NUM * A_DIM * HDIM) / 256, 256, 0, stream>>>(W2, w2t);

    layer0_kernel<<<B_ROWS / 64, 256, 0, stream>>>(cur_phase, feat, w0t, b0, sbuf);
    layer12_kernel<<<dim3(B_ROWS / 32, P_NUM), 256, 0, stream>>>(
        sbuf, w1t, b1, w2t, b2, cnt, offs, perm, out);
}
